// SampleNet_21371757265375
// MI455X (gfx1250) — matrix operand order repaired and checked by reading, not compiled, not run on hardware
//
#include <hip/hip_runtime.h>
#include <math.h>

typedef __attribute__((ext_vector_type(16))) _Float16 v16h;
typedef __attribute__((ext_vector_type(8)))  float    v8f;

#define Q 256
#define COND 128
#define A_CH 192
#define B_CH 32
#define BATCH 32
#define TLEN 2048
#define M_TOT (BATCH * TLEN)   // 65536 rows, ordered m = t*32 + b

// GRU-A dynamic LDS layout (CDNA5: up to 320KB per workgroup)
#define WHH_LDS_BYTES   (576 * 192 * 2)            // 221184: packed Whh_a, LDS-resident
#define HPACK_LDS_BYTES (2 * 6 * 32 * 16 * 2)      //  12288: h state, packed A layout
#define GHS_LDS_BYTES   (32 * 576 * 2)             //  36864: gate pre-activations (f16)
#define GRUA_LDS_BYTES  (WHH_LDS_BYTES + HPACK_LDS_BYTES + GHS_LDS_BYTES)  // 270336

// ---------------------------------------------------------------------------
// Packed operand layouts (CDNA5 WMMA f16 16x16x32, wave32) per ISA 7.12.2:
//  A (16x32 MxK):  lane = (m&15) + 16*((k>>3)&1), half = (k&7) | ((k>>4)<<3)
//  B (32x16 KxN):  lane = (n&15) + 16*((k>>3)&1), half = (k&7) | ((k>>4)<<3)
//  D (16x16 f32):  VGPR v, lanes 0-15 -> M=v,N=lane ; lanes 16-31 -> M=8+v,N=lane-16
// Buffers are [tileM/N][ktile][lane][16 halves] so each lane reads one v16h.
// ---------------------------------------------------------------------------
__device__ __forceinline__ int a_pack_idx(int m, int k, int KT) {
    int mt = m >> 4, kt = k >> 5, r = m & 15, kk = k & 31;
    int lane = r + (((kk >> 3) & 1) << 4);
    int half = (kk & 7) | ((kk >> 4) << 3);
    return (((mt * KT + kt) << 5) + lane) * 16 + half;
}
__device__ __forceinline__ int b_pack_idx(int n, int k, int KT) {
    int nt = n >> 4, kt = k >> 5, c = n & 15, kk = k & 31;
    int lane = c + (((kk >> 3) & 1) << 4);
    int half = (kk & 7) | ((kk >> 4) << 3);
    return (((nt * KT + kt) << 5) + lane) * 16 + half;
}

__device__ __forceinline__ float sigmoidf_(float x) { return 1.f / (1.f + expf(-x)); }

// --------------------------- weight packing (f32 -> f16, B layout) ----------
__global__ void pack_b_kernel(const float* __restrict__ w, _Float16* __restrict__ out,
                              int N, int K) {
    int idx = blockIdx.x * blockDim.x + threadIdx.x;
    if (idx >= N * K) return;
    int n = idx / K, k = idx - n * K;
    out[b_pack_idx(n, k, K >> 5)] = (_Float16)w[idx];
}

// ----------------- build x = [f, pe, se, ee] in packed A layout -------------
__global__ void build_x_kernel(const float* __restrict__ f, const float* __restrict__ p,
                               const float* __restrict__ s, const float* __restrict__ e,
                               const float* __restrict__ emb,
                               _Float16* __restrict__ Ax,   // [65536 x 896] packed, KT=28
                               _Float16* __restrict__ Axb)  // [65536 x 320] packed, KT=10 (f part: cols 192..319)
{
    int m = blockIdx.x;
    int t = m >> 5, b = m & 31;
    int bt = b * TLEN + t;
    float rp = p[bt] * 255.f, rs = s[bt] * 255.f, re = e[bt] * 255.f;
    int lp = min(max((int)floorf(rp), 0), 254);
    int ls = min(max((int)floorf(rs), 0), 254);
    int le = min(max((int)floorf(re), 0), 254);
    float fp = rp - (float)lp, fs = rs - (float)ls, fe = re - (float)le;
    const float* frow = f + (size_t)bt * COND;

    for (int k = threadIdx.x; k < 896; k += blockDim.x) {
        float val;
        if (k < COND) {
            val = frow[k];
            Axb[a_pack_idx(m, A_CH + k, 10)] = (_Float16)val;  // f part of GRU-B input
        } else {
            int seg = (k - COND) >> 8, c = (k - COND) & 255;
            int lo; float fr;
            if (seg == 0)      { lo = lp; fr = fp; }
            else if (seg == 1) { lo = ls; fr = fs; }
            else               { lo = le; fr = fe; }
            val = emb[lo * Q + c] * (1.f - fr) + emb[(lo + 1) * Q + c] * fr;
        }
        Ax[a_pack_idx(m, k, 28)] = (_Float16)val;
    }
}

// ------------------- generic packed WMMA GEMM: C[M,N] = A x B ---------------
// grid: (MT/8, NT/NW), block 256 (8 waves); each wave computes a 16 x (16*NW)
// output strip: one A fragment feeds NW WMMAs (A traffic / NW, 4:5 wmma:load).
template <int NW>
__global__ void wmma_gemm_kernel(const _Float16* __restrict__ A, const _Float16* __restrict__ B,
                                 float* __restrict__ C, int KT, int N) {
    int wave = threadIdx.x >> 5, lane = threadIdx.x & 31;
    int mtile = blockIdx.x * 8 + wave;
    int ntile0 = blockIdx.y * NW;
    const _Float16* Ab = A + ((size_t)mtile * KT * 512) + lane * 16;
    const _Float16* Bb = B + ((size_t)ntile0 * KT * 512) + lane * 16;
    v8f acc[NW];
#pragma unroll
    for (int i = 0; i < NW; ++i) acc[i] = (v8f){};
    for (int kt = 0; kt < KT; ++kt) {
        v16h av = *(const v16h*)(Ab + kt * 512);
#pragma unroll
        for (int i = 0; i < NW; ++i) {
            v16h bv = *(const v16h*)(Bb + ((size_t)i * KT + kt) * 512);
            acc[i] = __builtin_amdgcn_wmma_f32_16x16x32_f16(false, av, false, bv,
                                                            (short)0, acc[i], false, false);
        }
    }
    int mbase = mtile * 16 + ((lane >> 4) << 3);
    int nl = lane & 15;
#pragma unroll
    for (int i = 0; i < NW; ++i) {
        int n = (ntile0 + i) * 16 + nl;
        for (int v = 0; v < 8; ++v)
            C[(size_t)(mbase + v) * N + n] = acc[i][v];
    }
}

// ------------------------------ GRU-A recurrence ----------------------------
// Single workgroup, 24 waves. The full packed Whh (216 KB) is staged into the
// WGP's 320 KB LDS once, so the per-step 72-tile h @ Whh^T runs entirely out
// of LDS (no per-step VMEM/scratch reloads, low VGPR pressure). h also lives
// in LDS in packed A layout. Each wave owns a fixed mtile and 3 ntiles
// {nt, nt+12, nt+24}: one A fragment per k-block feeds 3 WMMAs.
__global__ void __launch_bounds__(768) gru_a_kernel(const float* __restrict__ gi,
                                                    const _Float16* __restrict__ Bhh, // N=576,K=192 packed
                                                    _Float16* __restrict__ Axb)       // KT=10, cols 0..191
{
    extern __shared__ char smem[];
    _Float16* whh   = (_Float16*)smem;                                        // 110592 halves
    _Float16* hpack = (_Float16*)(smem + WHH_LDS_BYTES);                      //   6144 halves
    _Float16* ghs   = (_Float16*)(smem + WHH_LDS_BYTES + HPACK_LDS_BYTES);    //  18432 halves
    int tid = threadIdx.x, wave = tid >> 5, lane = tid & 31;
    int mt = wave & 1, ntb = wave >> 1;           // wave -> (mtile, base ntile)

    // stage Whh into LDS (16B vectors), zero h state
    {
        const uint4* src = (const uint4*)Bhh;
        uint4* dst = (uint4*)whh;
        for (int i = tid; i < WHH_LDS_BYTES / 16; i += 768) dst[i] = src[i];
        for (int i = tid; i < HPACK_LDS_BYTES / 2; i += 768) hpack[i] = (_Float16)0.f;
    }
    __syncthreads();

    for (int t = 0; t < TLEN; ++t) {
        // phase 1: gh = h @ Whh^T   (3 ntiles per wave, shared A fragment, all LDS)
        {
            v8f acc[3];
#pragma unroll
            for (int i = 0; i < 3; ++i) acc[i] = (v8f){};
            for (int kb = 0; kb < 6; ++kb) {
                v16h av = *(const v16h*)(hpack + (((mt * 6 + kb) << 5) + lane) * 16);
#pragma unroll
                for (int i = 0; i < 3; ++i) {
                    int nt = ntb + i * 12;
                    v16h bv = *(const v16h*)(whh + (((nt * 6 + kb) << 5) + lane) * 16);
                    acc[i] = __builtin_amdgcn_wmma_f32_16x16x32_f16(false, av, false, bv,
                                                                    (short)0, acc[i], false, false);
                }
            }
            int mbase = mt * 16 + ((lane >> 4) << 3);
            int nl = lane & 15;
#pragma unroll
            for (int i = 0; i < 3; ++i) {
                int n = (ntb + i * 12) * 16 + nl;
                for (int v = 0; v < 8; ++v)
                    ghs[(mbase + v) * 576 + n] = (_Float16)acc[i][v];
            }
        }
        __syncthreads();

        // phase 2: gates + state update (6144 (b,j) pairs over 768 threads)
        const float* gir = gi + (size_t)t * 32 * 576;
        for (int i = 0; i < 8; ++i) {
            int idx = tid + i * 768;
            int b = idx / A_CH, j = idx - b * A_CH;
            const float* g = gir + b * 576;
            float hr = (float)ghs[b * 576 + j];
            float hz = (float)ghs[b * 576 + A_CH + j];
            float hn = (float)ghs[b * 576 + 2 * A_CH + j];
            float r = sigmoidf_(g[j] + hr);
            float z = sigmoidf_(g[A_CH + j] + hz);
            float nn = tanhf(g[2 * A_CH + j] + r * hn);
            // packed position of (row=b, col=j) in hpack
            int kk = j & 31;
            int pl = (b & 15) + (((kk >> 3) & 1) << 4);
            int ph = (kk & 7) | ((kk >> 4) << 3);
            int pidx = ((((b >> 4) * 6 + (j >> 5)) << 5) + pl) * 16 + ph;
            float hold = (float)hpack[pidx];
            float hnew = (1.f - z) * nn + z * hold;
            hpack[pidx] = (_Float16)hnew;
            Axb[a_pack_idx(t * 32 + b, j, 10)] = (_Float16)hnew;  // GRU-B input, h part
        }
        __syncthreads();
    }
}

// ------------------------------ GRU-B recurrence ----------------------------
// Per-wave Whh_b B fragment (32 B/lane) is reloaded from global Bhh on each step of the time
// loop (optimal -- do not disturb).
__global__ void __launch_bounds__(384) gru_b_kernel(const float* __restrict__ gi,
                                                    const _Float16* __restrict__ Bhh, // N=96,K=32 packed
                                                    _Float16* __restrict__ Ahb)       // [65536 x 32] packed, KT=1
{
    __shared__ _Float16 hpack[2 * 32 * 16];       // h [32x32] packed A layout
    __shared__ float ghs[32 * 96];
    int tid = threadIdx.x, wave = tid >> 5, lane = tid & 31;

    for (int i = tid; i < 2 * 32 * 16; i += 384) hpack[i] = (_Float16)0.f;
    __syncthreads();

    for (int t = 0; t < TLEN; ++t) {
        {   // phase 1: 12 tiles, one per wave, single k-block
            int mt = wave & 1, nt = wave >> 1;
            v16h av = *(const v16h*)(hpack + ((mt << 5) + lane) * 16);
            v16h bv = *(const v16h*)(Bhh + ((size_t)(nt << 5) + lane) * 16);
            v8f acc = {};
            acc = __builtin_amdgcn_wmma_f32_16x16x32_f16(false, av, false, bv,
                                                         (short)0, acc, false, false);
            int mbase = mt * 16 + ((lane >> 4) << 3);
            int n = nt * 16 + (lane & 15);
            for (int v = 0; v < 8; ++v)
                ghs[(mbase + v) * 96 + n] = acc[v];
        }
        __syncthreads();

        const float* gir = gi + (size_t)t * 32 * 96;
        for (int idx = tid; idx < 1024; idx += 384) {
            int b = idx >> 5, j = idx & 31;
            const float* g = gir + b * 96;
            float r = sigmoidf_(g[j] + ghs[b * 96 + j]);
            float z = sigmoidf_(g[B_CH + j] + ghs[b * 96 + B_CH + j]);
            float nn = tanhf(g[2 * B_CH + j] + r * ghs[b * 96 + 2 * B_CH + j]);
            int pl = (b & 15) + (((j >> 3) & 1) << 4);
            int ph = (j & 7) | ((j >> 4) << 3);
            int pidx = (((b >> 4) << 5) + pl) * 16 + ph;
            float hold = (float)hpack[pidx];
            float hnew = (1.f - z) * nn + z * hold;
            hpack[pidx] = (_Float16)hnew;
            Ahb[a_pack_idx(t * 32 + b, j, 1)] = (_Float16)hnew;
        }
        __syncthreads();
    }
}

// --------------- FC GEMM [65536x32]x[32x512] + fused epilogue ---------------
// out = tanh(hb @ fc_w^T + fc_b) * a, then pairwise channel sum -> [B,T,256].
// Adjacent N columns live in adjacent lanes of the D layout -> shfl_xor(1).
__global__ void fc_kernel(const _Float16* __restrict__ Ahb, const _Float16* __restrict__ Bfc,
                          const float* __restrict__ fc_b, const float* __restrict__ avec,
                          float* __restrict__ out) {
    int wave = threadIdx.x >> 5, lane = threadIdx.x & 31;
    int mtile = blockIdx.x * 8 + wave;
    int ntile = blockIdx.y;
    v16h av = *(const v16h*)(Ahb + ((size_t)(mtile << 5) + lane) * 16);
    v16h bv = *(const v16h*)(Bfc + ((size_t)(ntile << 5) + lane) * 16);
    v8f acc = {};
    acc = __builtin_amdgcn_wmma_f32_16x16x32_f16(false, av, false, bv,
                                                 (short)0, acc, false, false);
    int n = ntile * 16 + (lane & 15);
    float bias = fc_b[n], scale = avec[n];
    int mbase = mtile * 16 + ((lane >> 4) << 3);
    for (int v = 0; v < 8; ++v) {
        float val = tanhf(acc[v] + bias) * scale;
        float partner = __shfl_xor(val, 1, 32);
        if (!(lane & 1)) {
            int m = mbase + v;
            int b = m & 31, t = m >> 5;
            out[((size_t)b * TLEN + t) * Q + (n >> 1)] = val + partner;
        }
    }
}

// ---------------------------------------------------------------------------
extern "C" void kernel_launch(void* const* d_in, const int* in_sizes, int n_in,
                              void* d_out, int out_size, void* d_ws, size_t ws_size,
                              hipStream_t stream) {
    const float* f     = (const float*)d_in[0];
    const float* p     = (const float*)d_in[1];
    const float* sprev = (const float*)d_in[2];
    const float* eprev = (const float*)d_in[3];
    const float* emb   = (const float*)d_in[4];
    const float* wiha  = (const float*)d_in[5];
    const float* whha  = (const float*)d_in[6];
    const float* wihb  = (const float*)d_in[7];
    const float* whhb  = (const float*)d_in[8];
    const float* avec  = (const float*)d_in[9];
    const float* fcw   = (const float*)d_in[10];
    const float* fcb   = (const float*)d_in[11];
    float* out = (float*)d_out;

    char* ws = (char*)d_ws;
    size_t off = 0;
    auto alloc = [&](size_t bytes) { void* pt = ws + off; off = (off + bytes + 255) & ~(size_t)255; return pt; };
    _Float16* Ax   = (_Float16*)alloc((size_t)M_TOT * 896 * 2);   // x packed
    float*    giA  = (float*)   alloc((size_t)M_TOT * 576 * 4);   // GRU-A input proj
    _Float16* Axb  = (_Float16*)alloc((size_t)M_TOT * 320 * 2);   // [h_a, f] packed
    float*    giB  = (float*)   alloc((size_t)M_TOT * 96 * 4);    // GRU-B input proj
    _Float16* Ahb  = (_Float16*)alloc((size_t)M_TOT * 32 * 2);    // h_b packed
    _Float16* Bwia = (_Float16*)alloc((size_t)576 * 896 * 2);
    _Float16* Bwha = (_Float16*)alloc((size_t)576 * 192 * 2);
    _Float16* Bwib = (_Float16*)alloc((size_t)96 * 320 * 2);
    _Float16* Bwhb = (_Float16*)alloc((size_t)96 * 32 * 2);
    _Float16* Bfc  = (_Float16*)alloc((size_t)512 * 32 * 2);
    (void)ws_size; (void)in_sizes; (void)n_in; (void)out_size;

    // allow >64KB dynamic LDS for the GRU-A kernel (CDNA5: 320KB per WG).
    // Non-enqueuing call; graph-capture safe; result intentionally ignored.
    (void)hipFuncSetAttribute((const void*)gru_a_kernel,
                              hipFuncAttributeMaxDynamicSharedMemorySize, GRUA_LDS_BYTES);

    // pack weights into WMMA B layouts
    pack_b_kernel<<<(576 * 896 + 255) / 256, 256, 0, stream>>>(wiha, Bwia, 576, 896);
    pack_b_kernel<<<(576 * 192 + 255) / 256, 256, 0, stream>>>(whha, Bwha, 576, 192);
    pack_b_kernel<<<(96 * 320 + 255) / 256, 256, 0, stream>>>(wihb, Bwib, 96, 320);
    pack_b_kernel<<<(96 * 32 + 255) / 256, 256, 0, stream>>>(whhb, Bwhb, 96, 32);
    pack_b_kernel<<<(512 * 32 + 255) / 256, 256, 0, stream>>>(fcw, Bfc, 512, 32);

    // embeddings + concat -> packed A matrices
    build_x_kernel<<<M_TOT, 128, 0, stream>>>(f, p, sprev, eprev, emb, Ax, Axb);

    // gi_a = x @ w_ih_a^T   (M=65536, N=576, K=896); 16x64 strip per wave
    wmma_gemm_kernel<4><<<dim3(M_TOT / 128, 576 / 64), 256, 0, stream>>>(Ax, Bwia, giA, 28, 576);

    // GRU-A recurrence (sequential over T; single WGP-resident workgroup,
    // Whh LDS-resident via 270KB dynamic LDS)
    gru_a_kernel<<<1, 768, GRUA_LDS_BYTES, stream>>>(giA, Bwha, Axb);

    // gi_b = [h_a, f] @ w_ih_b^T   (M=65536, N=96, K=320); 16x32 strip per wave
    wmma_gemm_kernel<2><<<dim3(M_TOT / 128, 96 / 32), 256, 0, stream>>>(Axb, Bwib, giB, 10, 96);

    // GRU-B recurrence
    gru_b_kernel<<<1, 384, 0, stream>>>(giB, Bwhb, Ahb);

    // FC + tanh*a + pairwise sum -> d_out [32,2048,256]
    fc_kernel<<<dim3(M_TOT / 128, 512 / 16), 256, 0, stream>>>(Ahb, Bfc, fcb, avec, out);
}
